// FuseMoE_45114336477544
// MI455X (gfx1250) — matrix operand order repaired and checked by reading, not compiled, not run on hardware
//
#include <hip/hip_runtime.h>
#include <hip/hip_bf16.h>
#include <math.h>

#define D_MODEL 512
#define N_EXP   8
#define D_HID   2048
#define MT      64      // token rows per expert tile (4 x 16-row m-halves)
#define NH      (MT/16)
#define FC      128     // Dh chunk (hidden columns per phase)
#define SA_STR  520     // LDS row stride (bf16) for h tile: 1040B, 16B aligned, bank-spread
#define SP_STR  136     // LDS row stride (bf16) for hid chunk: 272B
#define SA_BYTES (MT * SA_STR * 2)
#define SP_BYTES (MT * SP_STR * 2)
// sA + double-buffered sP + tok/gate
#define SMEM_BYTES (SA_BYTES + 2 * SP_BYTES + MT * 4 + MT * 4)

typedef __attribute__((ext_vector_type(16))) __bf16 v16bf;
typedef __attribute__((ext_vector_type(8)))  __bf16 v8bf;
typedef __attribute__((ext_vector_type(4)))  __bf16 v4bf;
typedef __attribute__((ext_vector_type(8)))  float  v8f;

// ---------------- WMMA fragment helpers (CDNA5 wave32 layouts) ----------------

__device__ __forceinline__ v16bf frag_cat(v8bf lo, v8bf hi) {
  return __builtin_shufflevector(lo, hi, 0,1,2,3,4,5,6,7,8,9,10,11,12,13,14,15);
}

// A (16-bit, 16x32): lane holds row M=lane%16; K = kbase+{0..7} and kbase+16+{0..7},
// kbase=(lane/16)*8. Caller passes p = row_base + kk + kbase (16B aligned).
__device__ __forceinline__ v16bf load_fragA(const __bf16* p) {
  v8bf lo = *(const v8bf*)(p);
  v8bf hi = *(const v8bf*)(p + 16);
  return frag_cat(lo, hi);
}

// B (16-bit, 32x16): lane holds col N=lane%16; K = kbase+{0..7} and kbase+16+{0..7},
// kbase=(lane/16)*8. Caller passes p = col_base + kk + kbase (16B aligned).
__device__ __forceinline__ v16bf load_fragB(const __bf16* p) {
  v8bf lo = *(const v8bf*)(p);
  v8bf hi = *(const v8bf*)(p + 16);
  return frag_cat(lo, hi);
}

__device__ __forceinline__ v8f wmma_bf16(v16bf a, v16bf b, v8f c) {
  // (neg_a, A, neg_b, B, c_mod, C, reuse_a, reuse_b)
  return __builtin_amdgcn_wmma_f32_16x16x32_bf16(false, a, false, b, (short)0, c,
                                                 false, false);
}

// Branchless erf, Abramowitz-Stegun 7.1.26 (max abs err 1.5e-7).
// Single-instruction v_rcp_f32 / v_exp_f32 instead of libm's ~30-op erff.
__device__ __forceinline__ float erf_fast(float x) {
  float ax = fabsf(x);
  float t  = __builtin_amdgcn_rcpf(fmaf(0.3275911f, ax, 1.0f));
  float p  = fmaf(1.061405429f, t, -1.453152027f);
  p = fmaf(p, t, 1.421413741f);
  p = fmaf(p, t, -0.284496736f);
  p = fmaf(p, t, 0.254829592f);
  p = p * t;
  float r = fmaf(-p, __expf(-ax * ax), 1.0f);
  return copysignf(r, x);
}

__device__ __forceinline__ float gelu_exact(float v) {
  return 0.5f * v * (1.0f + erf_fast(v * 0.70710678118654752f));
}

// ---------------- Kernel 0: zero per-expert counters ----------------

__global__ void zero_counts_kernel(int* __restrict__ counts) {
  if (threadIdx.x < N_EXP) counts[threadIdx.x] = 0;
}

// ---------------- Kernel 1: fp32 [R][C] -> bf16 [C][R] per expert ----------------

__global__ void transpose_cvt_kernel(const float* __restrict__ src,
                                     __bf16* __restrict__ dst, int R, int C) {
  __shared__ float tile[32][33];
  const size_t esz = (size_t)R * C;
  src += (size_t)blockIdx.z * esz;
  dst += (size_t)blockIdx.z * esz;
  int c0 = blockIdx.x * 32, r0 = blockIdx.y * 32;
  int tx = threadIdx.x, ty = threadIdx.y;       // 32 x 8
#pragma unroll
  for (int i = 0; i < 4; i++) {
    int r = ty * 4 + i;
    tile[r][tx] = src[(size_t)(r0 + r) * C + c0 + tx];   // coalesced read
  }
  __syncthreads();
#pragma unroll
  for (int i = 0; i < 4; i++) {
    int c = ty * 4 + i;
    dst[(size_t)(c0 + c) * R + r0 + tx] = (__bf16)tile[tx][c];  // coalesced write
  }
}

// ---------------- Kernel 2: LayerNorm + router top-2 + bucketing ----------------

__global__ void ln_router_kernel(const float* __restrict__ x,
                                 const float* __restrict__ lng,
                                 const float* __restrict__ lnb,
                                 const float* __restrict__ wr,
                                 const float* __restrict__ br,
                                 __bf16* __restrict__ hbf,
                                 float* __restrict__ out,
                                 int* __restrict__ counts,
                                 int* __restrict__ atok,
                                 float* __restrict__ agate, int N) {
  __shared__ float sWr[N_EXP * D_MODEL];
  for (int i = threadIdx.x; i < N_EXP * D_MODEL; i += blockDim.x) sWr[i] = wr[i];
  __syncthreads();

  const int lane = threadIdx.x & 31;
  const int tok  = blockIdx.x * (blockDim.x >> 5) + (threadIdx.x >> 5);
  if (tok >= N) return;

  const float* xp = x + (size_t)tok * D_MODEL;
  float4 xv[4];
  float s = 0.f, s2 = 0.f;
#pragma unroll
  for (int j = 0; j < 4; j++) {
    xv[j] = *(const float4*)(xp + j * 128 + lane * 4);
    s  += xv[j].x + xv[j].y + xv[j].z + xv[j].w;
    s2 += xv[j].x * xv[j].x + xv[j].y * xv[j].y + xv[j].z * xv[j].z + xv[j].w * xv[j].w;
  }
#pragma unroll
  for (int m = 16; m >= 1; m >>= 1) {
    s  += __shfl_xor(s,  m, 32);
    s2 += __shfl_xor(s2, m, 32);
  }
  const float mu = s * (1.0f / D_MODEL);
  const float rs = rsqrtf(s2 * (1.0f / D_MODEL) - mu * mu + 1e-5f);

  float acc[N_EXP];
#pragma unroll
  for (int e = 0; e < N_EXP; e++) acc[e] = 0.f;

#pragma unroll
  for (int j = 0; j < 4; j++) {
    const float* xf = (const float*)&xv[j];
    v4bf hb;
#pragma unroll
    for (int c = 0; c < 4; c++) {
      int d = j * 128 + lane * 4 + c;
      float h = (xf[c] - mu) * rs * lng[d] + lnb[d];
      hb[c] = (__bf16)h;
#pragma unroll
      for (int e = 0; e < N_EXP; e++) acc[e] += h * sWr[e * D_MODEL + d];
    }
    *(v4bf*)(hbf + (size_t)tok * D_MODEL + j * 128 + lane * 4) = hb;        // A feed
    *(float4*)(out + (size_t)tok * D_MODEL + j * 128 + lane * 4) = xv[j];   // residual
  }
#pragma unroll
  for (int e = 0; e < N_EXP; e++) {
#pragma unroll
    for (int m = 16; m >= 1; m >>= 1) acc[e] += __shfl_xor(acc[e], m, 32);
    acc[e] += br[e];
  }

  // top-2 (first occurrence wins on ties, matching lax.top_k)
  float l0 = -INFINITY, l1 = -INFINITY;
  int i0 = 0, i1 = 0;
#pragma unroll
  for (int e = 0; e < N_EXP; e++) {
    float le = acc[e];
    if (le > l0) { l1 = l0; i1 = i0; l0 = le; i0 = e; }
    else if (le > l1) { l1 = le; i1 = e; }
  }
  float e1 = __expf(l1 - l0);
  float inv = 1.0f / (1.0f + e1);
  float g0 = inv, g1 = e1 * inv;

  if (lane == 0) {
    int p0 = atomicAdd(&counts[i0], 1);
    atok[(size_t)i0 * N + p0]  = tok;
    agate[(size_t)i0 * N + p0] = g0;
    int p1 = atomicAdd(&counts[i1], 1);
    atok[(size_t)i1 * N + p1]  = tok;
    agate[(size_t)i1 * N + p1] = g1;
  }
}

// ---------------- Kernel 3: fused expert FFN (grouped GEMM, WMMA bf16) ----------------
// Block = 256 threads (8 waves) = one expert x 64 token rows. 64-row M-tile amortizes the
// W1[e]+W2[e] L2 stream over 4 WMMAs per loaded B fragment. sP double-buffered: one
// barrier per Dh-chunk. Dynamic LDS (~102KB; gfx1250 allows 320KB/workgroup).

__global__ void __launch_bounds__(256)
expert_ffn_kernel(const __bf16* __restrict__ hbf,
                  const __bf16* __restrict__ W1T,   // [E][Dh][D]  (col-major B, GEMM1)
                  const __bf16* __restrict__ W2T,   // [E][D][Dh]  (col-major B, GEMM2)
                  const float* __restrict__ b1,
                  const float* __restrict__ b2,
                  const int* __restrict__ counts,
                  const int* __restrict__ atok,
                  const float* __restrict__ agate,
                  float* __restrict__ out, int N) {
  const int e = blockIdx.y;
  int cnt = __builtin_amdgcn_readfirstlane(counts[e]);
  const int base = blockIdx.x * MT;
  if (base >= cnt) return;

  extern __shared__ char smem[];
  __bf16 (*sA)[SA_STR]  = (__bf16(*)[SA_STR])(smem);
  __bf16 (*sP0)[SP_STR] = (__bf16(*)[SP_STR])(smem + SA_BYTES);
  __bf16 (*sP1)[SP_STR] = (__bf16(*)[SP_STR])(smem + SA_BYTES + SP_BYTES);
  int*   sTok  = (int*)(smem + SA_BYTES + 2 * SP_BYTES);
  float* sGate = (float*)(smem + SA_BYTES + 2 * SP_BYTES + MT * 4);

  const int tid = threadIdx.x;
  if (tid < MT) {
    int ai = base + tid;
    int tk = (ai < cnt) ? atok[(size_t)e * N + ai] : -1;
    sTok[tid]  = tk;
    sGate[tid] = (ai < cnt) ? agate[(size_t)e * N + ai] : 0.f;
  }
  __syncthreads();

  // stage h tile: MT rows x 512 bf16; 8 segments of 64 cols per row
  for (int idx = tid; idx < MT * 8; idx += 256) {
    int r = idx >> 3;
    int cseg = (idx & 7) * 64;
    int tk = sTok[r];
    if (tk >= 0) {
      const __bf16* src = hbf + (size_t)tk * D_MODEL + cseg;
#pragma unroll
      for (int i = 0; i < 8; i++)
        *(v8bf*)&sA[r][cseg + i * 8] = *(const v8bf*)(src + i * 8);
    } else {
#pragma unroll
      for (int i = 0; i < 64; i++) sA[r][cseg + i] = (__bf16)0.f;
    }
  }
  __syncthreads();

  const int lane  = tid & 31;
  const int wv    = tid >> 5;          // wave 0..7
  const int col16 = lane & 15;
  const int half  = lane >> 4;
  const int kbA   = half * 8;          // A-fragment K base
  const int kbB   = half * 8;          // B-fragment K base

  v8f acc[NH][4];
#pragma unroll
  for (int h = 0; h < NH; h++)
#pragma unroll
    for (int t = 0; t < 4; t++)
#pragma unroll
      for (int v = 0; v < 8; v++) acc[h][t][v] = 0.f;

  const __bf16* W1e = W1T + (size_t)e * D_HID * D_MODEL;
  const __bf16* W2e = W2T + (size_t)e * D_MODEL * D_HID;

  for (int ch = 0; ch < D_HID; ch += FC) {
    __bf16 (*sP)[SP_STR] = ((ch >> 7) & 1) ? sP1 : sP0;   // ping-pong buffer

    // ---- GEMM1: wave wv computes hid cols [ch + wv*16, +16) for all MT rows ----
    const int ncol = ch + wv * 16 + col16;
    const __bf16* bcol = W1e + (size_t)ncol * D_MODEL;   // K contiguous per column
    v8f p[NH];
#pragma unroll
    for (int h = 0; h < NH; h++)
#pragma unroll
      for (int v = 0; v < 8; v++) p[h][v] = 0.f;
#pragma unroll
    for (int kk = 0; kk < D_MODEL; kk += 32) {
      v16bf bf = load_fragB(bcol + kk + kbB);
      v16bf af[NH];
#pragma unroll
      for (int h = 0; h < NH; h++)                 // batch all A loads: one DS clause,
        af[h] = load_fragA(&sA[col16 + 16 * h][kk + kbA]);
#pragma unroll
      for (int h = 0; h < NH; h++)                 // then drain with staggered dscnt waits
        p[h] = wmma_bf16(af[h], bf, p[h]);
    }
    const float bias1 = b1[e * D_HID + ncol];

    // Safe without a pre-write barrier: this chunk writes buffer ch%2 while any
    // straggler reads buffer (ch-1)%2; the single post-write barrier prevents lapping.
#pragma unroll
    for (int h = 0; h < NH; h++)
#pragma unroll
      for (int v = 0; v < 8; v++) {
        int m0 = v + half * 8 + 16 * h;            // C layout: row = vgpr + 8*(lane/16)
        sP[m0][wv * 16 + col16] = (__bf16)gelu_exact(p[h][v] + bias1);
      }
    __syncthreads();

    // ---- GEMM2: wave wv accumulates out cols [wv*64, +64) over this chunk ----
#pragma unroll
    for (int kk2 = 0; kk2 < FC; kk2 += 32) {
      v16bf a2[NH];
#pragma unroll
      for (int h = 0; h < NH; h++)
        a2[h] = load_fragA(&sP[col16 + 16 * h][kk2 + kbA]);
#pragma unroll
      for (int t = 0; t < 4; t++) {
        int n = wv * 64 + t * 16 + col16;
        v16bf bf = load_fragB(W2e + (size_t)n * D_HID + ch + kk2 + kbB);
#pragma unroll
        for (int h = 0; h < NH; h++)
          acc[h][t] = wmma_bf16(a2[h], bf, acc[h][t]);
      }
    }
  }

  // ---- epilogue: +b2, gate, atomic add into residual ----
  float bias2[4];
#pragma unroll
  for (int t = 0; t < 4; t++)
    bias2[t] = b2[e * D_MODEL + wv * 64 + t * 16 + col16];

#pragma unroll
  for (int h = 0; h < NH; h++)
#pragma unroll
    for (int v = 0; v < 8; v++) {
      int m = v + half * 8 + 16 * h;
      int tk = sTok[m];
      if (tk < 0) continue;
      float g = sGate[m];
      float* orow = out + (size_t)tk * D_MODEL + wv * 64 + col16;
#pragma unroll
      for (int t = 0; t < 4; t++)
        atomicAdd(orow + t * 16, g * (acc[h][t][v] + bias2[t]));
    }
}

// ---------------- host launcher ----------------

extern "C" void kernel_launch(void* const* d_in, const int* in_sizes, int n_in,
                              void* d_out, int out_size, void* d_ws, size_t ws_size,
                              hipStream_t stream) {
  const float* x   = (const float*)d_in[0];
  const float* lng = (const float*)d_in[1];
  const float* lnb = (const float*)d_in[2];
  const float* wr  = (const float*)d_in[3];
  const float* br  = (const float*)d_in[4];
  const float* W1  = (const float*)d_in[5];
  const float* b1  = (const float*)d_in[6];
  const float* W2  = (const float*)d_in[7];
  const float* b2  = (const float*)d_in[8];
  float* out = (float*)d_out;
  const int N = in_sizes[0] / D_MODEL;   // 16384 tokens

  char* ws = (char*)d_ws;
  size_t off = 0;
  __bf16* W1T = (__bf16*)(ws + off); off += (size_t)N_EXP * D_HID * D_MODEL * 2;
  __bf16* W2T = (__bf16*)(ws + off); off += (size_t)N_EXP * D_MODEL * D_HID * 2;
  __bf16* hbf = (__bf16*)(ws + off); off += (size_t)N * D_MODEL * 2;
  int*   counts = (int*)(ws + off);  off += 256;
  int*   atokp  = (int*)(ws + off);  off += (size_t)N_EXP * N * sizeof(int);
  float* agatep = (float*)(ws + off);

  zero_counts_kernel<<<1, 64, 0, stream>>>(counts);
  // W1 [E][512][2048] -> W1T [E][2048][512]
  transpose_cvt_kernel<<<dim3(D_HID / 32, D_MODEL / 32, N_EXP), dim3(32, 8), 0, stream>>>(
      W1, W1T, D_MODEL, D_HID);
  // W2 [E][2048][512] -> W2T [E][512][2048]
  transpose_cvt_kernel<<<dim3(D_MODEL / 32, D_HID / 32, N_EXP), dim3(32, 8), 0, stream>>>(
      W2, W2T, D_HID, D_MODEL);
  ln_router_kernel<<<N / 8, 256, 0, stream>>>(x, lng, lnb, wr, br, hbf, out,
                                              counts, atokp, agatep, N);
  expert_ffn_kernel<<<dim3((N + MT - 1) / MT, N_EXP), 256, SMEM_BYTES, stream>>>(
      hbf, W1T, W2T, b1, b2, counts, atokp, agatep, out, N);
}
